// MaaSGraphNetwork_55035710931345
// MI455X (gfx1250) — compile-verified, matrix operand order checked by reading, not run on hardware
//
#include <hip/hip_runtime.h>

#define NN 100000
#define NE 1600000
#define IN_DIM 5
#define HID 64
#define HEADS 4
#define C2 256          // HEADS*HID
#define OUT_DIM 4
#define NEG_SLOPE 0.2f
#define EPS 1e-5f
#define ENC_NEG_INF 0x007FFFFFu   // encF(-inf)

typedef __attribute__((ext_vector_type(16))) __bf16    v16bf;
typedef __attribute__((ext_vector_type(8)))  float     v8f;
typedef __attribute__((ext_vector_type(4)))  unsigned  u32x4;
typedef __attribute__((ext_vector_type(4)))  float     f32x4;

union V16U { u32x4 q[2]; v16bf v; };

static __device__ __forceinline__ unsigned short f2bf(float f) {
  unsigned u = __float_as_uint(f);
  unsigned r = u + 0x7FFFu + ((u >> 16) & 1u);   // round-to-nearest-even
  return (unsigned short)(r >> 16);
}
// order-preserving float<->uint encoding for atomic max
static __device__ __forceinline__ unsigned encF(float f) {
  unsigned u = __float_as_uint(f);
  return (u & 0x80000000u) ? ~u : (u | 0x80000000u);
}
static __device__ __forceinline__ float decF(unsigned k) {
  return (k & 0x80000000u) ? __uint_as_float(k & 0x7FFFFFFFu) : __uint_as_float(~k);
}

// ---------------- init fills ----------------
__global__ void fill_f32(float* p, float v, int n) {
  int i = blockIdx.x * blockDim.x + threadIdx.x;
  if (i < n) p[i] = v;
}
__global__ void fill_u32(unsigned* p, unsigned v, int n) {
  int i = blockIdx.x * blockDim.x + threadIdx.x;
  if (i < n) p[i] = v;
}

// ---------------- edge aggregation: deg, sum x[src], sum edge_attr ----------------
__global__ void edge_agg(const int* __restrict__ src, const int* __restrict__ dst,
                         const float* __restrict__ x, const float* __restrict__ ea,
                         float* deg, float* agg, float* easum) {
  int e = blockIdx.x * blockDim.x + threadIdx.x;
  if (e >= NE) return;
  int s = src[e], d = dst[e];
  atomicAdd(&deg[d], 1.0f);
#pragma unroll
  for (int k = 0; k < IN_DIM; k++) atomicAdd(&agg[d * IN_DIM + k], x[s * IN_DIM + k]);
  atomicAdd(&easum[d * 2 + 0], ea[e * 2 + 0]);
  atomicAdd(&easum[d * 2 + 1], ea[e * 2 + 1]);
}

// ---------------- SAGEConv: h = (agg/deg)@Wl^T + bl + x@Wr^T ----------------
__global__ void sage(const float* __restrict__ x, const float* __restrict__ agg,
                     const float* __restrict__ deg, const float* __restrict__ wl,
                     const float* __restrict__ bl, const float* __restrict__ wr,
                     float* h1) {
  int gid = blockIdx.x * blockDim.x + threadIdx.x;
  if (gid >= NN * HID) return;
  int n = gid / HID, c = gid % HID;
  float dv = deg[n]; dv = dv > 1.f ? dv : 1.f;
  float inv = 1.f / dv;
  float acc = bl[c];
#pragma unroll
  for (int k = 0; k < IN_DIM; k++)
    acc += agg[n * IN_DIM + k] * inv * wl[c * IN_DIM + k] + x[n * IN_DIM + k] * wr[c * IN_DIM + k];
  h1[gid] = acc;
}

// ---------------- channel-wise sum & sum-of-squares over rows ----------------
__global__ void reduce_sum_sq(const float* __restrict__ in, int nrows, int C,
                              float* gsum, float* gsq) {
  __shared__ float ss[256];
  __shared__ float qq[256];
  int tid = threadIdx.x;
  int gid = blockIdx.x * blockDim.x + tid;
  int T = gridDim.x * blockDim.x;
  int ch = gid % C;
  float s = 0.f, q = 0.f;
  for (int r = gid / C; r < nrows; r += T / C) {
    float v = in[(size_t)r * C + ch];
    s += v; q += v * v;
  }
  ss[tid] = s; qq[tid] = q;
  __syncthreads();
  for (int off = blockDim.x >> 1; off >= C; off >>= 1) {
    if (tid < off) { ss[tid] += ss[tid + off]; qq[tid] += qq[tid + off]; }
    __syncthreads();
  }
  if (tid < C) { atomicAdd(&gsum[tid], ss[tid]); atomicAdd(&gsq[tid], qq[tid]); }
}

// mean/rstd for GraphNorm over y = in + prebias
__global__ void norm_finalize(const float* gsum, const float* gsq, const float* prebias,
                              const float* ms, int C, float invn, float* mean, float* rstd) {
  int c = threadIdx.x;
  if (c >= C) return;
  float b = prebias ? prebias[c] : 0.f;
  float eg = gsum[c] * invn;          // E[g]
  float m = eg + b;                   // E[y]
  float ey2 = gsq[c] * invn + 2.f * b * eg + b * b;
  float msv = ms[c];
  float var = ey2 - (2.f * msv - msv * msv) * m * m;
  mean[c] = m;
  rstd[c] = rsqrtf(var + EPS);
}

// apply GraphNorm + ELU; write f32 and/or bf16
__global__ void norm_apply_elu(const float* __restrict__ in, const float* __restrict__ prebias,
                               const float* __restrict__ w, const float* __restrict__ b,
                               const float* __restrict__ ms, const float* __restrict__ mean,
                               const float* __restrict__ rstd, int C, int total,
                               float* out_f32, unsigned short* out_bf) {
  int gid = blockIdx.x * blockDim.x + threadIdx.x;
  if (gid >= total) return;
  int c = gid % C;
  float v = in[gid] + (prebias ? prebias[c] : 0.f);
  v = w[c] * (v - ms[c] * mean[c]) * rstd[c] + b[c];
  v = v > 0.f ? v : (__expf(v) - 1.f);           // ELU
  if (out_f32) out_f32[gid] = v;
  if (out_bf) out_bf[gid] = f2bf(v);
}

__global__ void conv_bf16(const float* __restrict__ in, unsigned short* out, int n) {
  int i = blockIdx.x * blockDim.x + threadIdx.x;
  if (i < n) out[i] = f2bf(in[i]);
}

// ---------------- WMMA GEMM: xp[N,256] = h[N,64](bf16) @ gat_w[256,64]^T(bf16) ----------------
// One wave: 16 rows x 64 cols (4 col-tiles), K=64 in two 32-steps. 25000 waves exactly.
__global__ void gat_gemm_wmma(const unsigned short* __restrict__ A,
                              const unsigned short* __restrict__ B,
                              float* __restrict__ xp) {
  int wave = (blockIdx.x * blockDim.x + threadIdx.x) >> 5;
  int lane = threadIdx.x & 31;
  int rt = wave >> 2;                 // 0..6249
  int cg = (wave & 3) * 64;           // 0,64,128,192
  int l15 = lane & 15;
  int khalf = (lane >> 4) * 8;        // lanes 0-15 -> K {0..7,16..23}; 16-31 -> {8..15,24..31}
  // A fragments (row = rt*16 + l15, 64 bf16 per row = 8 x u32x4)
  const u32x4* pa = reinterpret_cast<const u32x4*>(A + (size_t)(rt * 16 + l15) * HID);
  V16U a0, a1;
  a0.q[0] = pa[(khalf) >> 3];
  a0.q[1] = pa[(khalf + 16) >> 3];
  a1.q[0] = pa[(32 + khalf) >> 3];
  a1.q[1] = pa[(32 + khalf + 16) >> 3];
  int rbase = rt * 16 + ((lane >> 4) << 3);
#pragma unroll
  for (int ct = 0; ct < 4; ct++) {
    int col = cg + ct * 16 + l15;
    const u32x4* pb = reinterpret_cast<const u32x4*>(B + (size_t)col * HID);
    V16U b0, b1;
    b0.q[0] = pb[(khalf) >> 3];
    b0.q[1] = pb[(khalf + 16) >> 3];
    b1.q[0] = pb[(32 + khalf) >> 3];
    b1.q[1] = pb[(32 + khalf + 16) >> 3];
    v8f c = {0.f, 0.f, 0.f, 0.f, 0.f, 0.f, 0.f, 0.f};
    c = __builtin_amdgcn_wmma_f32_16x16x32_bf16(false, a0.v, false, b0.v, (short)0, c, false, false);
    c = __builtin_amdgcn_wmma_f32_16x16x32_bf16(false, a1.v, false, b1.v, (short)0, c, false, false);
    // C layout: VGPR r holds D[M = r (+8 for hi lanes)][N = lane&15]
#pragma unroll
    for (int r = 0; r < 8; r++)
      xp[(size_t)(rbase + r) * C2 + col] = c[r];
  }
}

// ---------------- per-node attention scores a_src, a_dst (LDS-staged, 32 nodes/block) ------
// LDS layout: row stride 264, head stride 66 -> compute-phase banks (8*nl+2*hd+j)%64 distinct.
__global__ void node_attn(const float* __restrict__ xp, const float* __restrict__ att_src,
                          const float* __restrict__ att_dst, float* asrc, float* adst) {
  __shared__ float sx[32 * 264];
  __shared__ float sas[C2];
  __shared__ float sad[C2];
  int tid = threadIdx.x;
  int n0 = blockIdx.x * 32;
  int nrows = NN - n0; if (nrows > 32) nrows = 32;
  if (tid < C2) { sas[tid] = att_src[tid]; sad[tid] = att_dst[tid]; }
  for (int i = tid; i < nrows * C2; i += 256) {
    int row = i >> 8, col = i & 255;
    sx[row * 264 + (col >> 6) * 66 + (col & 63)] = xp[(size_t)n0 * C2 + i];
  }
  __syncthreads();
  int nl = tid >> 2, hd = tid & 3;
  if (nl < nrows) {
    const float* row = sx + nl * 264 + hd * 66;
    const float* ws_ = sas + hd * HID;
    const float* wd_ = sad + hd * HID;
    float s = 0.f, d = 0.f;
#pragma unroll 8
    for (int j = 0; j < HID; j++) { float v = row[j]; s += v * ws_[j]; d += v * wd_[j]; }
    asrc[(n0 + nl) * 4 + hd] = s;
    adst[(n0 + nl) * 4 + hd] = d;
  }
}

// w2[head][c] = sum_j gat_we[head*64+j][c] * att_edge[head][j]  (collapses ep entirely)
__global__ void edge_w2(const float* __restrict__ gat_we, const float* __restrict__ att_edge,
                        float* w2) {
  int t = threadIdx.x;
  if (t >= HEADS * 2) return;
  int hd = t >> 1, c = t & 1;
  float s = 0.f;
  for (int j = 0; j < HID; j++) s += gat_we[(hd * HID + j) * 2 + c] * att_edge[hd * HID + j];
  w2[hd * 2 + c] = s;
}

// ---------------- edge pass 1: alpha + segment-max ----------------
__global__ void edge_alpha(const int* __restrict__ src, const int* __restrict__ dst,
                           const float* __restrict__ ea, const float* __restrict__ easum,
                           const float* __restrict__ deg, const float* __restrict__ asrc,
                           const float* __restrict__ adst, const float* __restrict__ w2,
                           float* alpha, unsigned* amax) {
  int e = blockIdx.x * blockDim.x + threadIdx.x;
  if (e >= NE + NN) return;
  int s, d; float e0, e1;
  if (e < NE) {
    s = src[e]; d = dst[e]; e0 = ea[e * 2]; e1 = ea[e * 2 + 1];
  } else {
    int n = e - NE; s = d = n;
    float dv = deg[n]; dv = dv > 1.f ? dv : 1.f;
    e0 = easum[n * 2] / dv; e1 = easum[n * 2 + 1] / dv;
  }
#pragma unroll
  for (int hd = 0; hd < HEADS; hd++) {
    float a = asrc[s * 4 + hd] + adst[d * 4 + hd] + e0 * w2[hd * 2] + e1 * w2[hd * 2 + 1];
    a = a > 0.f ? a : NEG_SLOPE * a;               // leaky relu
    alpha[(size_t)e * 4 + hd] = a;
    atomicMax(&amax[d * 4 + hd], encF(a));
  }
}

// ---------------- edge pass 2: exp(alpha - max), segment-sum ----------------
__global__ void edge_exp(const int* __restrict__ dst, float* alpha,
                         const unsigned* __restrict__ amax, float* denom) {
  int gid = blockIdx.x * blockDim.x + threadIdx.x;
  if (gid >= (NE + NN) * HEADS) return;
  int e = gid >> 2, hd = gid & 3;
  int d = (e < NE) ? dst[e] : (e - NE);
  float v = __expf(alpha[gid] - decF(amax[d * 4 + hd]));
  alpha[gid] = v;                                  // in place -> expa
  atomicAdd(&denom[d * 4 + hd], v);
}

// ---------------- edge pass 3: scatter messages (coalesced float4 chunks) ----------------
// 16 consecutive lanes cover the 64 channels of one (edge, head): 1 b128 gather + 4 atomics.
__global__ void edge_msg(const int* __restrict__ src, const int* __restrict__ dst,
                         const float* __restrict__ expa, const float* __restrict__ denom,
                         const float* __restrict__ xp, float* g) {
  int gid = blockIdx.x * blockDim.x + threadIdx.x;
  if (gid >= (NE + NN) * HEADS * 16) return;
  int chunk = gid & 15;             // which float4 of the 64 channels
  int eh = gid >> 4;
  int e = eh >> 2, hd = eh & 3;
  int s, d;
  if (e < NE) { s = src[e]; d = dst[e]; } else { s = d = e - NE; }
  float at = expa[eh] / denom[d * 4 + hd];
  const f32x4 v = *reinterpret_cast<const f32x4*>(xp + (size_t)s * C2 + hd * HID + chunk * 4);
  float* gr = g + (size_t)d * C2 + hd * HID + chunk * 4;
  atomicAdd(&gr[0], v[0] * at);
  atomicAdd(&gr[1], v[1] * at);
  atomicAdd(&gr[2], v[2] * at);
  atomicAdd(&gr[3], v[3] * at);
}

// ---------------- output head: relu(g2 @ out_w^T + out_b), LDS-staged ----------------
__global__ void out_lin(const float* __restrict__ g2, const float* __restrict__ ow,
                        const float* __restrict__ ob, float* out) {
  __shared__ float sg[32 * 260];
  __shared__ float sw[4 * C2];
  int tid = threadIdx.x;
  int n0 = blockIdx.x * 32;
  int nrows = NN - n0; if (nrows > 32) nrows = 32;
  for (int i = tid; i < 4 * C2; i += 256) sw[i] = ow[i];
  for (int i = tid; i < nrows * C2; i += 256)
    sg[(i >> 8) * 260 + (i & 255)] = g2[(size_t)n0 * C2 + i];
  __syncthreads();
  int nl = tid >> 2, o = tid & 3;
  if (nl < nrows) {
    const float* r = sg + nl * 260;
    const float* w = sw + o * C2;
    float a = ob[o];
#pragma unroll 8
    for (int k = 0; k < C2; k++) a += r[k] * w[k];
    out[(n0 + nl) * 4 + o] = a > 0.f ? a : 0.f;
  }
}

extern "C" void kernel_launch(void* const* d_in, const int* in_sizes, int n_in,
                              void* d_out, int out_size, void* d_ws, size_t ws_size,
                              hipStream_t stream) {
  (void)in_sizes; (void)n_in; (void)out_size; (void)ws_size;
  const float* x        = (const float*)d_in[0];
  const float* ea       = (const float*)d_in[1];
  const float* sage_wl  = (const float*)d_in[2];
  const float* sage_bl  = (const float*)d_in[3];
  const float* sage_wr  = (const float*)d_in[4];
  const float* n1_w     = (const float*)d_in[5];
  const float* n1_b     = (const float*)d_in[6];
  const float* n1_ms    = (const float*)d_in[7];
  const float* gat_w    = (const float*)d_in[8];
  const float* att_src  = (const float*)d_in[9];
  const float* att_dst  = (const float*)d_in[10];
  const float* att_edge = (const float*)d_in[11];
  const float* gat_we   = (const float*)d_in[12];
  const float* gat_b    = (const float*)d_in[13];
  const float* n2_w     = (const float*)d_in[14];
  const float* n2_b     = (const float*)d_in[15];
  const float* n2_ms    = (const float*)d_in[16];
  const float* out_w    = (const float*)d_in[17];
  const float* out_b    = (const float*)d_in[18];
  const int*   eidx     = (const int*)d_in[19];
  const int* src = eidx;
  const int* dst = eidx + NE;
  float* out = (float*)d_out;

  // workspace carve-up (~283 MB; g2 reuses xp)
  char* base = (char*)d_ws;
  size_t off = 0;
  auto take = [&](size_t bytes) -> void* {
    off = (off + 255) & ~(size_t)255;
    void* p = base + off; off += bytes; return p;
  };
  float* deg    = (float*)take((size_t)NN * 4);
  float* agg    = (float*)take((size_t)NN * IN_DIM * 4);
  float* easum  = (float*)take((size_t)NN * 2 * 4);
  float* h1     = (float*)take((size_t)NN * HID * 4);
  unsigned short* hbf = (unsigned short*)take((size_t)NN * HID * 2);
  unsigned short* wbf = (unsigned short*)take((size_t)C2 * HID * 2);
  float* xp     = (float*)take((size_t)NN * C2 * 4);
  float* asrc   = (float*)take((size_t)NN * HEADS * 4);
  float* adst   = (float*)take((size_t)NN * HEADS * 4);
  float* w2     = (float*)take(8 * 4);
  float* alpha  = (float*)take((size_t)(NE + NN) * HEADS * 4);
  unsigned* amax= (unsigned*)take((size_t)NN * HEADS * 4);
  float* denom  = (float*)take((size_t)NN * HEADS * 4);
  float* gacc   = (float*)take((size_t)NN * C2 * 4);
  float* sum1 = (float*)take(64 * 4);  float* sq1 = (float*)take(64 * 4);
  float* mean1= (float*)take(64 * 4);  float* rstd1=(float*)take(64 * 4);
  float* sum2 = (float*)take(256 * 4); float* sq2 = (float*)take(256 * 4);
  float* mean2= (float*)take(256 * 4); float* rstd2=(float*)take(256 * 4);
  float* g2 = xp;   // xp dead after edge_msg

  const int B = 256;
  auto gr = [](long n) { return (int)((n + 255) / 256); };

  // init
  fill_f32<<<gr(NN), B, 0, stream>>>(deg, 0.f, NN);
  fill_f32<<<gr((long)NN * IN_DIM), B, 0, stream>>>(agg, 0.f, NN * IN_DIM);
  fill_f32<<<gr((long)NN * 2), B, 0, stream>>>(easum, 0.f, NN * 2);
  fill_f32<<<gr((long)NN * HEADS), B, 0, stream>>>(denom, 0.f, NN * HEADS);
  fill_f32<<<gr((long)NN * C2), B, 0, stream>>>(gacc, 0.f, NN * C2);
  fill_f32<<<1, B, 0, stream>>>(sum1, 0.f, 64);
  fill_f32<<<1, B, 0, stream>>>(sq1, 0.f, 64);
  fill_f32<<<1, B, 0, stream>>>(sum2, 0.f, 256);
  fill_f32<<<1, B, 0, stream>>>(sq2, 0.f, 256);
  fill_u32<<<gr((long)NN * HEADS), B, 0, stream>>>(amax, ENC_NEG_INF, NN * HEADS);

  // SAGE layer
  edge_agg<<<gr(NE), B, 0, stream>>>(src, dst, x, ea, deg, agg, easum);
  sage<<<gr((long)NN * HID), B, 0, stream>>>(x, agg, deg, sage_wl, sage_bl, sage_wr, h1);

  // GraphNorm1 + ELU -> bf16 h
  reduce_sum_sq<<<512, B, 0, stream>>>(h1, NN, HID, sum1, sq1);
  norm_finalize<<<1, 64, 0, stream>>>(sum1, sq1, nullptr, n1_ms, HID, 1.f / NN, mean1, rstd1);
  norm_apply_elu<<<gr((long)NN * HID), B, 0, stream>>>(h1, nullptr, n1_w, n1_b, n1_ms,
                                                       mean1, rstd1, HID, NN * HID,
                                                       nullptr, hbf);

  // GAT projection via WMMA bf16
  conv_bf16<<<gr(C2 * HID), B, 0, stream>>>(gat_w, wbf, C2 * HID);
  gat_gemm_wmma<<<3125, B, 0, stream>>>(hbf, wbf, xp);   // 25000 waves exactly

  // attention
  node_attn<<<3125, B, 0, stream>>>(xp, att_src, att_dst, asrc, adst);  // 32 nodes/block
  edge_w2<<<1, 64, 0, stream>>>(gat_we, att_edge, w2);
  edge_alpha<<<gr((long)(NE + NN)), B, 0, stream>>>(src, dst, ea, easum, deg, asrc, adst,
                                                    w2, alpha, amax);
  edge_exp<<<gr((long)(NE + NN) * HEADS), B, 0, stream>>>(dst, alpha, amax, denom);
  edge_msg<<<gr((long)(NE + NN) * HEADS * 16), B, 0, stream>>>(src, dst, alpha, denom, xp, gacc);

  // GraphNorm2(+gat_b) + ELU -> g2 (reuses xp)
  reduce_sum_sq<<<512, B, 0, stream>>>(gacc, NN, C2, sum2, sq2);
  norm_finalize<<<1, 256, 0, stream>>>(sum2, sq2, gat_b, n2_ms, C2, 1.f / NN, mean2, rstd2);
  norm_apply_elu<<<gr((long)NN * C2), B, 0, stream>>>(gacc, gat_b, n2_w, n2_b, n2_ms,
                                                      mean2, rstd2, C2, NN * C2,
                                                      g2, nullptr);

  // output head
  out_lin<<<3125, B, 0, stream>>>(g2, out_w, out_b, out);
}
